// CGCNN_29102698398262
// MI455X (gfx1250) — matrix operand order repaired and checked by reading, not compiled, not run on hardware
//
#include <hip/hip_runtime.h>
#include <hip/hip_bf16.h>

// ---------------------------------------------------------------------------
// CGCNN forward for MI455X (gfx1250, wave32).
// Edge-message GEMMs (80 GFLOP dominant cost) run on v_wmma_f32_16x16x32_bf16.
// B (weight) fragments are register-resident per wave: each wave owns one
// 16-wide N-tile and grid-strides over edge tiles, so the inner loop is only
// h-row gathers + 10 WMMAs + gate/scatter.
// ---------------------------------------------------------------------------

typedef __attribute__((ext_vector_type(16))) __bf16 v16bf;
typedef __attribute__((ext_vector_type(8)))  __bf16 v8bf;
typedef __attribute__((ext_vector_type(8)))  float  v8f;

#define F_IN 92
#define DDIM 64
#define ZDIM 131
#define LNUM 3
#define KT_N 5   // ceil(160/32) K-tiles (Z padded 131 -> 160)
#define NT_N 4   // 64 / 16 N-tiles

// ---------------------------------------------------------------- zero -----
__global__ void zero_kernel(float* p, long n) {
    long i = (long)blockIdx.x * blockDim.x + threadIdx.x;
    if (i < n) p[i] = 0.0f;
}

// -------------------------------------------------------------- pre FC -----
// h = relu(x @ pre_W + pre_b); also emit bf16 copy for WMMA A operands.
__global__ void pre_fc_kernel(const float* __restrict__ x,
                              const float* __restrict__ W,
                              const float* __restrict__ b,
                              float* __restrict__ h, __bf16* __restrict__ hb,
                              int N_) {
    __shared__ float xs[F_IN];
    int n = blockIdx.x;
    int d = threadIdx.x;            // 64 threads
    if (n >= N_) return;
    for (int k = d; k < F_IN; k += DDIM) xs[k] = x[(long)n * F_IN + k];
    __syncthreads();
    float acc = b[d];
#pragma unroll 4
    for (int k = 0; k < F_IN; ++k) acc = fmaf(xs[k], W[k * DDIM + d], acc);
    acc = fmaxf(acc, 0.0f);
    h[(long)n * DDIM + d]  = acc;
    hb[(long)n * DDIM + d] = (__bf16)acc;
}

// ----------------------------------------------------------- in-degree -----
__global__ void deg_kernel(const int* __restrict__ ei, float* __restrict__ deg,
                           int E_) {
    long e = (long)blockIdx.x * blockDim.x + threadIdx.x;
    if (e < E_) atomicAdd(&deg[ei[(long)E_ + e]], 1.0f);   // dst = edge_index[1]
}

// --------------------------------------------------- pack B fragments ------
// wpack fragment idx = ((layer*2 + mat)*KT_N + kt)*NT_N + nt, 512 bf16 each,
// stored [lane][elem].  B layout (32x16 bf16): lanes 0-15 -> col=lane,
// K = kt*32 + e; lanes 16-31 -> col = lane-16, K = kt*32 + 16 + e.
__global__ void pack_weights_kernel(const float* __restrict__ Wf,
                                    const float* __restrict__ Ws,
                                    __bf16* __restrict__ wpack) {
    int idx  = blockIdx.x;          // 0 .. L*2*KT_N*NT_N-1
    int nt   = idx & 3;
    int kt   = (idx >> 2) % KT_N;
    int lm   = idx / (KT_N * NT_N);
    int mat  = lm & 1;
    int l    = lm >> 1;
    int lane = threadIdx.x;         // 32
    int colL = lane & 15;
    int halfB = lane >> 4;
    const float* W = mat ? Ws : Wf;
    __bf16* dst = wpack + (long)idx * 512 + lane * 16;
#pragma unroll
    for (int e = 0; e < 16; ++e) {
        int K   = kt * 32 + halfB * 16 + e;
        int col = nt * 16 + colL;
        float v = (K < ZDIM) ? W[((long)l * ZDIM + K) * DDIM + col] : 0.0f;
        dst[e] = (__bf16)v;
    }
}

// ------------------------------------------------------- edge messages -----
// Wave-stationary B: wave w owns ntile (w & 3), keeps its 10 weight
// fragments in VGPRs, and grid-strides over 16-edge tiles.
// A = z-tile [16 x 32] bf16 per ktile (contiguous k = 16*half + e, as in B pack),
// C/D = f32 accumulators initialized with bias.
// msg = sigmoid(zWf+bf) * softplus(zWs+bs); scatter-add into agg[dst].
__global__ __launch_bounds__(256)
void edge_msg_kernel(const int* __restrict__ ei,
                     const float* __restrict__ eattr,
                     const __bf16* __restrict__ hb,
                     const __bf16* __restrict__ wpack,
                     const float* __restrict__ bfv_,
                     const float* __restrict__ bsv_,
                     float* __restrict__ agg,
                     int layer, int E_, int n_tiles) {
    const int wavesPerBlock = blockDim.x >> 5;
    const int w    = blockIdx.x * wavesPerBlock + (threadIdx.x >> 5);
    const int lane = threadIdx.x & 31;
    const int nt      = w & 3;
    const int tstart  = w >> 2;
    const int tstride = (gridDim.x * wavesPerBlock) >> 2;
    const int half = lane >> 4;
    const int m    = lane & 15;
    const int col  = nt * 16 + m;            // C/D: col = lane & 15

    const float bf0 = bfv_[layer * DDIM + col];
    const float bs0 = bsv_[layer * DDIM + col];

    // ---- register-resident B fragments (held for the whole kernel) ----
    v16bf wf[KT_N], wsg[KT_N];
#pragma unroll
    for (int kt = 0; kt < KT_N; ++kt) {
        long fb = ((long)((layer * 2 + 0) * KT_N + kt) * NT_N + nt) * 512;
        long sb = ((long)((layer * 2 + 1) * KT_N + kt) * NT_N + nt) * 512;
        wf[kt]  = *(const v16bf*)(wpack + fb + lane * 16);
        wsg[kt] = *(const v16bf*)(wpack + sb + lane * 16);
    }

    for (int t = tstart; t < n_tiles; t += tstride) {
        long e  = (long)t * 16 + m;
        bool ev = (e < E_);
        long ec = ev ? e : (long)E_ - 1;
        int  si = ei[ec];                    // src  = edge_index[0]
        int  di = ei[(long)E_ + ec];         // dst  = edge_index[1]

        // ---- A fragments (contiguous k order: k = 16*half + e) ----
        v16bf a[KT_N];
#pragma unroll
        for (int kt = 0; kt < 4; ++kt) {
            int row = (kt < 2) ? di : si;    // z = [h_dst | h_src | e_attr]
            const __bf16* rp = hb + (long)row * DDIM;
            int cb0 = (kt & 1) * 32 + half * 16;
            v8bf lo = *(const v8bf*)(rp + cb0);
            v8bf hi = *(const v8bf*)(rp + cb0 + 8);
#pragma unroll
            for (int i = 0; i < 8; ++i) { a[kt][i] = lo[i]; a[kt][i + 8] = hi[i]; }
        }
        {   // ktile 4: K 128..130 = edge_attr, rest zero padding
            v16bf tt;
#pragma unroll
            for (int i = 0; i < 16; ++i) tt[i] = (__bf16)0.0f;
            if (half == 0 && ev) {
                tt[0] = (__bf16)eattr[ec * 3 + 0];
                tt[1] = (__bf16)eattr[ec * 3 + 1];
                tt[2] = (__bf16)eattr[ec * 3 + 2];
            }
            a[4] = tt;
        }

        // ---- 10 back-to-back WMMAs, bias-initialized accumulators ----
        v8f cf, cs;
#pragma unroll
        for (int v = 0; v < 8; ++v) { cf[v] = bf0; cs[v] = bs0; }
#pragma unroll
        for (int kt = 0; kt < KT_N; ++kt) {
            cf = __builtin_amdgcn_wmma_f32_16x16x32_bf16(
                     false, a[kt], false, wf[kt], (short)0, cf, false, false);
            cs = __builtin_amdgcn_wmma_f32_16x16x32_bf16(
                     false, a[kt], false, wsg[kt], (short)0, cs, false, false);
        }

        // gate * softplus, scatter-add per output row M = v + 8*half
#pragma unroll
        for (int v = 0; v < 8; ++v) {
            int  M  = v + 8 * half;
            long ee = (long)t * 16 + M;
            int  r  = __shfl(di, M, 32);     // dst index held by lane M
            if (ee < E_) {
                float g  = 1.0f / (1.0f + __expf(-cf[v]));
                float xx = cs[v];
                float sp = fmaxf(xx, 0.0f) + __logf(1.0f + __expf(-fabsf(xx)));
                atomicAdd(&agg[(long)r * DDIM + col], g * sp);
            }
        }
    }
}

// ------------------------------------------- h += agg/deg + BN statistics --
__global__ void add_agg_kernel(float* __restrict__ h,
                               const float* __restrict__ agg,
                               const float* __restrict__ deg,
                               float* __restrict__ bnsum, int N_) {
    int d = threadIdx.x;                    // 64 threads, d = channel
    int base = blockIdx.x * 128;
    float s = 0.0f, s2 = 0.0f;
    for (int i = 0; i < 128; ++i) {
        int n = base + i;
        if (n < N_) {
            float dg = fmaxf(deg[n], 1.0f);
            long  ix = (long)n * DDIM + d;
            float v  = h[ix] + agg[ix] / dg;
            h[ix] = v;
            s += v; s2 += v * v;
        }
    }
    atomicAdd(&bnsum[d], s);
    atomicAdd(&bnsum[DDIM + d], s2);
}

// ------------------------------------------------------------- BN apply ----
__global__ void bn_apply_kernel(float* __restrict__ h, __bf16* __restrict__ hb,
                                float* __restrict__ agg,
                                const float* __restrict__ bnsum,
                                const float* __restrict__ gamma,
                                const float* __restrict__ beta,
                                int layer, int N_) {
    long i = (long)blockIdx.x * blockDim.x + threadIdx.x;
    if (i >= (long)N_ * DDIM) return;
    int d = (int)(i & (DDIM - 1));
    float inv_n = 1.0f / (float)N_;
    float mu  = bnsum[d] * inv_n;
    float var = bnsum[DDIM + d] * inv_n - mu * mu;
    float y = gamma[layer * DDIM + d] * (h[i] - mu) * rsqrtf(var + 1e-5f)
            + beta[layer * DDIM + d];
    h[i]  = y;
    hb[i] = (__bf16)y;
    agg[i] = 0.0f;                           // reset for next layer
}

// ----------------------------------------------------------- mean pool -----
__global__ void pool_kernel(const float* __restrict__ h,
                            const int* __restrict__ batch,
                            float* __restrict__ pooled, float* __restrict__ cnt,
                            int N_) {
    long i = (long)blockIdx.x * blockDim.x + threadIdx.x;
    if (i >= (long)N_ * DDIM) return;
    int n = (int)(i >> 6);
    int d = (int)(i & (DDIM - 1));
    int b = batch[n];
    atomicAdd(&pooled[(long)b * DDIM + d], h[i]);
    if (d == 0) atomicAdd(&cnt[b], 1.0f);
}

// ----------------------------------------------------------------- head ----
__global__ void head_kernel(const float* __restrict__ pooled,
                            const float* __restrict__ cnt,
                            const float* __restrict__ postW,
                            const float* __restrict__ postb,
                            const float* __restrict__ outW,
                            const float* __restrict__ outb,
                            float* __restrict__ out, int G_) {
    for (int g = threadIdx.x; g < G_; g += blockDim.x) {
        float pm[DDIM];
        float c = fmaxf(cnt[g], 1.0f);
#pragma unroll
        for (int d = 0; d < DDIM; ++d) pm[d] = pooled[(long)g * DDIM + d] / c;
        float o = outb[0];
        for (int j = 0; j < DDIM; ++j) {
            float t = postb[j];
#pragma unroll 8
            for (int d = 0; d < DDIM; ++d)
                t = fmaf(pm[d], postW[d * DDIM + j], t);
            o = fmaf(fmaxf(t, 0.0f), outW[j], o);
        }
        out[g] = o;
    }
}

// ---------------------------------------------------------------------------
extern "C" void kernel_launch(void* const* d_in, const int* in_sizes, int n_in,
                              void* d_out, int out_size, void* d_ws, size_t ws_size,
                              hipStream_t stream) {
    const float* x      = (const float*)d_in[0];
    const int*   ei     = (const int*)  d_in[1];
    const float* eattr  = (const float*)d_in[2];
    const int*   batch  = (const int*)  d_in[3];
    const float* preW   = (const float*)d_in[4];
    const float* preB   = (const float*)d_in[5];
    const float* convWf = (const float*)d_in[6];
    const float* convBf = (const float*)d_in[7];
    const float* convWs = (const float*)d_in[8];
    const float* convBs = (const float*)d_in[9];
    const float* bnG    = (const float*)d_in[10];
    const float* bnB    = (const float*)d_in[11];
    const float* postW  = (const float*)d_in[12];
    const float* postB  = (const float*)d_in[13];
    const float* outW   = (const float*)d_in[14];
    const float* outB   = (const float*)d_in[15];
    float* out = (float*)d_out;

    const int N_ = in_sizes[0] / F_IN;
    const int E_ = in_sizes[1] / 2;
    const int G_ = out_size;

    // workspace carve-up (256B aligned)
    auto align256 = [](size_t v) { return (v + 255) & ~(size_t)255; };
    char* ws = (char*)d_ws;
    size_t off = 0;
    float*  h      = (float*)(ws + off); off = align256(off + (size_t)N_ * DDIM * 4);
    float*  agg    = (float*)(ws + off); off = align256(off + (size_t)N_ * DDIM * 4);
    __bf16* hb     = (__bf16*)(ws + off); off = align256(off + (size_t)N_ * DDIM * 2);
    __bf16* wpack  = (__bf16*)(ws + off); off = align256(off + (size_t)LNUM * 2 * KT_N * NT_N * 512 * 2);
    float*  deg    = (float*)(ws + off); off = align256(off + (size_t)N_ * 4);
    float*  bnsum  = (float*)(ws + off); off = align256(off + 2 * DDIM * 4);
    float*  pooled = (float*)(ws + off); off = align256(off + (size_t)G_ * DDIM * 4);
    float*  cnt    = (float*)(ws + off); off = align256(off + (size_t)G_ * 4);
    (void)ws_size; (void)n_in;

    const int ZB = 256;
    auto zblocks = [&](long n) { return (int)((n + ZB - 1) / ZB); };

    // init
    zero_kernel<<<zblocks((long)N_ * DDIM), ZB, 0, stream>>>(agg, (long)N_ * DDIM);
    zero_kernel<<<zblocks(N_), ZB, 0, stream>>>(deg, N_);
    zero_kernel<<<1, ZB, 0, stream>>>(bnsum, 2 * DDIM);
    zero_kernel<<<zblocks((long)G_ * DDIM), ZB, 0, stream>>>(pooled, (long)G_ * DDIM);
    zero_kernel<<<1, ZB, 0, stream>>>(cnt, G_);

    // pre FC + degree + weight packing
    pre_fc_kernel<<<N_, DDIM, 0, stream>>>(x, preW, preB, h, hb, N_);
    deg_kernel<<<(E_ + ZB - 1) / ZB, ZB, 0, stream>>>(ei, deg, E_);
    pack_weights_kernel<<<LNUM * 2 * KT_N * NT_N, 32, 0, stream>>>(convWf, convWs, wpack);

    // CGConv layers
    const int n_tiles = (E_ + 15) / 16;
    const int eblocks = 1024;                        // 8192 waves -> 2048/ntile
    const int aab  = (N_ + 127) / 128;
    const int nb64 = (int)(((long)N_ * DDIM + ZB - 1) / ZB);
    for (int l = 0; l < LNUM; ++l) {
        edge_msg_kernel<<<eblocks, 256, 0, stream>>>(
            ei, eattr, hb, wpack, convBf, convBs, agg, l, E_, n_tiles);
        add_agg_kernel<<<aab, DDIM, 0, stream>>>(h, agg, deg, bnsum, N_);
        bn_apply_kernel<<<nb64, ZB, 0, stream>>>(h, hb, agg, bnsum, bnG, bnB, l, N_);
        zero_kernel<<<1, ZB, 0, stream>>>(bnsum, 2 * DDIM);
    }

    // pool + head
    pool_kernel<<<nb64, ZB, 0, stream>>>(h, batch, pooled, cnt, N_);
    head_kernel<<<1, 128, 0, stream>>>(pooled, cnt, postW, postB, outW, outB, out, G_);
}
